// CausalSelfAttention_16037407884068
// MI455X (gfx1250) — matrix operand order repaired and checked by reading, not compiled, not run on hardware
//
#include <hip/hip_runtime.h>
#include <cstdint>

typedef __attribute__((ext_vector_type(16))) __bf16 v16bf;
typedef __attribute__((ext_vector_type(8)))  __bf16 v8bf;
typedef __attribute__((ext_vector_type(8)))  float  v8f;

#define BDIM 256
static constexpr int T_SEQ = 4096;
static constexpr int CDIM  = 768;
static constexpr int HEADS = 12;
static constexpr int DH    = 64;

__device__ __forceinline__ unsigned short f2bf(float f) {
  unsigned int u = __builtin_bit_cast(unsigned int, f);
  u += 0x7FFFu + ((u >> 16) & 1u);   // round-to-nearest-even
  return (unsigned short)(u >> 16);
}

__device__ __forceinline__ v8f vzero8() {
  v8f z;
  #pragma unroll
  for (int i = 0; i < 8; ++i) z[i] = 0.0f;
  return z;
}

// Build a 16-element bf16 fragment from two 16-byte aligned 8-element loads.
__device__ __forceinline__ v16bf mk16(const unsigned short* p0, const unsigned short* p1) {
  v8bf lo = *(const v8bf*)p0;
  v8bf hi = *(const v8bf*)p1;
  return __builtin_shufflevector(lo, hi, 0,1,2,3,4,5,6,7,8,9,10,11,12,13,14,15);
}

// VALU cross-lane xor-shuffle within 16-lane groups via v_permlane16_b32
// (keeps softmax reductions off the DS pipe, which feeds WMMA fragments).
__device__ __forceinline__ float pl16(float v, unsigned s0, unsigned s1) {
  unsigned u = __builtin_bit_cast(unsigned, v);
  unsigned r = __builtin_amdgcn_permlane16(u, u, s0, s1, false, false);
  return __builtin_bit_cast(float, r);
}
__device__ __forceinline__ float rmax16(float v) {
  v = fmaxf(v, pl16(v, 0x67452301u, 0xEFCDAB89u));  // xor 1
  v = fmaxf(v, pl16(v, 0x54761032u, 0xDCFE98BAu));  // xor 2
  v = fmaxf(v, pl16(v, 0x32107654u, 0xBA98FEDCu));  // xor 4
  v = fmaxf(v, pl16(v, 0xFEDCBA98u, 0x76543210u));  // xor 8
  return v;
}
__device__ __forceinline__ float rsum16(float v) {
  v += pl16(v, 0x67452301u, 0xEFCDAB89u);
  v += pl16(v, 0x54761032u, 0xDCFE98BAu);
  v += pl16(v, 0x32107654u, 0xBA98FEDCu);
  v += pl16(v, 0xFEDCBA98u, 0x76543210u);
  return v;
}

__global__ void convert_bf16(const float* __restrict__ src, unsigned short* __restrict__ dst, int n) {
  int i = blockIdx.x * blockDim.x + threadIdx.x;
  if (i < n) dst[i] = f2bf(src[i]);
}

// dst[n*K + k] = src[k*N + n]  (store weights transposed so B fragments are K-contiguous)
__global__ void convert_transpose_bf16(const float* __restrict__ src, unsigned short* __restrict__ dst,
                                       int K, int N) {
  int i = blockIdx.x * blockDim.x + threadIdx.x;
  if (i < K * N) {
    int k = i / N, n = i - k * N;
    dst[(size_t)n * K + k] = f2bf(src[i]);
  }
}

// C[M,N] = A[M,K] * B[K,N] + bias, with A bf16 row-major and Bt = B^T bf16 row-major.
// Double-buffered LDS fed by global_load_async_to_lds_b128 (ASYNCcnt pipeline).
// EPI==0: scatter to Q/K/V [B,H,T,Dh] bf16 (Q scaled by 1/sqrt(Dh)); EPI==1: fp32 out.
template <int EPI>
__global__ __launch_bounds__(BDIM) void gemm_bf16(
    const unsigned short* __restrict__ A,
    const unsigned short* __restrict__ Bt,
    const float* __restrict__ bias,
    float* __restrict__ Cout,
    unsigned short* __restrict__ Qo,
    unsigned short* __restrict__ Ko,
    unsigned short* __restrict__ Vo,
    int M, int N, int K) {
  __shared__ __align__(16) unsigned short lA[2][128 * 40];
  __shared__ __align__(16) unsigned short lB[2][128 * 40];

  const int tid = threadIdx.x;
  const int lane = tid & 31, wid = tid >> 5;
  const int half = lane >> 4, lanen = lane & 15;
  const int wm = wid & 3, wn = wid >> 2;          // 4x2 wave grid, each wave 32x64
  const int m0 = blockIdx.y * 128, n0 = blockIdx.x * 128;

  // Issue async copies of one 128x32 A tile and one 128x32 Bt tile into buffer `buf`.
  auto issue_tile = [&](int buf, int k0) {
    #pragma unroll
    for (int p = 0; p < 2; ++p) {
      int chunk = p * 256 + tid;
      int row = chunk >> 2;
      int c8 = (chunk & 3) * 8;
      unsigned la = (unsigned)(uintptr_t)&lA[buf][row * 40 + c8];
      unsigned lb = (unsigned)(uintptr_t)&lB[buf][row * 40 + c8];
      unsigned long long ga = (unsigned long long)(uintptr_t)&A[(size_t)(m0 + row) * K + k0 + c8];
      unsigned long long gb = (unsigned long long)(uintptr_t)&Bt[(size_t)(n0 + row) * K + k0 + c8];
      asm volatile("global_load_async_to_lds_b128 %0, %1, off" :: "v"(la), "v"(ga) : "memory");
      asm volatile("global_load_async_to_lds_b128 %0, %1, off" :: "v"(lb), "v"(gb) : "memory");
    }
  };

  v8f acc[2][4];
  #pragma unroll
  for (int i = 0; i < 2; ++i)
    #pragma unroll
    for (int j = 0; j < 4; ++j) acc[i][j] = vzero8();

  const int nk = K / 32;
  issue_tile(0, 0);
  for (int kk = 0; kk < nk; ++kk) {
    asm volatile("s_wait_asynccnt 0" ::: "memory");   // this wave's tile kk is in LDS
    __syncthreads();                                  // everyone's tile kk visible;
                                                      // all readers of buf (kk+1)&1 are done
    if (kk + 1 < nk) issue_tile((kk + 1) & 1, (kk + 1) * 32);  // prefetch overlaps compute

    const unsigned short* bufA = lA[kk & 1];
    const unsigned short* bufB = lB[kk & 1];
    v16bf af[2], bfm[4];
    #pragma unroll
    for (int mt = 0; mt < 2; ++mt) {            // A: lane<16 -> K {0..7,16..23}; +8 for upper half
      const unsigned short* pr = &bufA[(wm * 32 + mt * 16 + lanen) * 40 + half * 8];
      af[mt] = mk16(pr, pr + 16);
    }
    #pragma unroll
    for (int nt = 0; nt < 4; ++nt) {            // B: lane<16 -> K {0..7,16..23}; +8 for upper half
      const unsigned short* pr = &bufB[(wn * 64 + nt * 16 + lanen) * 40 + half * 8];
      bfm[nt] = mk16(pr, pr + 16);
    }
    #pragma unroll
    for (int mt = 0; mt < 2; ++mt)
      #pragma unroll
      for (int nt = 0; nt < 4; ++nt)
        acc[mt][nt] = __builtin_amdgcn_wmma_f32_16x16x32_bf16(
            false, af[mt], false, bfm[nt], (short)0, acc[mt][nt], false, false);
  }

  #pragma unroll
  for (int mt = 0; mt < 2; ++mt) {
    #pragma unroll
    for (int nt = 0; nt < 4; ++nt) {
      int col = n0 + wn * 64 + nt * 16 + lanen;
      float bv = bias[col];
      if constexpr (EPI == 0) {
        int sel = col / CDIM;                    // 0=q 1=k 2=v
        int rem = col - sel * CDIM;
        int h = rem >> 6, dd = rem & 63;
        unsigned short* dst = sel == 0 ? Qo : (sel == 1 ? Ko : Vo);
        float scl = (sel == 0) ? 0.125f : 1.0f;  // fold 1/sqrt(64) into Q
        #pragma unroll
        for (int r = 0; r < 8; ++r) {
          int m = m0 + wm * 32 + mt * 16 + r + half * 8;
          int bb = m >> 12, t = m & (T_SEQ - 1);
          dst[((size_t)(bb * HEADS + h) * T_SEQ + t) * DH + dd] = f2bf((acc[mt][nt][r] + bv) * scl);
        }
      } else {
        #pragma unroll
        for (int r = 0; r < 8; ++r) {
          int m = m0 + wm * 32 + mt * 16 + r + half * 8;
          Cout[(size_t)m * N + col] = acc[mt][nt][r] + bv;
        }
      }
    }
  }
}

// Flash attention: one block = 128 query rows of one (b,h); each wave owns 16 rows.
__global__ __launch_bounds__(BDIM) void flash_attn(
    const unsigned short* __restrict__ Qg,
    const unsigned short* __restrict__ Kg,
    const unsigned short* __restrict__ Vg,
    unsigned short* __restrict__ Yg) {
  __shared__ __align__(16) unsigned short ldsK[64 * 72];    // K tile  [key][d]
  __shared__ __align__(16) unsigned short ldsVT[64 * 72];   // V tile transposed [d][key]
  __shared__ __align__(16) unsigned short ldsP[8 * 16 * 72];// per-wave P transpose buffer

  const int tid = threadIdx.x;
  const int lane = tid & 31, wid = tid >> 5;
  const int half = lane >> 4, lanen = lane & 15;
  const int bh = blockIdx.y;
  const int qs = blockIdx.x * 128;
  const size_t base = (size_t)bh * T_SEQ * DH;

  // Q fragments (A-matrix layout) for this wave's 16 rows, kept in registers
  const unsigned short* qp = Qg + base + (size_t)(qs + wid * 16 + lanen) * DH;
  v16bf aq[2];
  #pragma unroll
  for (int c = 0; c < 2; ++c) {
    int o1 = c * 32 + half * 8;
    aq[c] = mk16(qp + o1, qp + o1 + 16);
  }

  float mrow[8], lrow[8];
  #pragma unroll
  for (int r = 0; r < 8; ++r) { mrow[r] = -1e30f; lrow[r] = 0.0f; }
  v8f o[4];
  #pragma unroll
  for (int d = 0; d < 4; ++d) o[d] = vzero8();

  unsigned short* pw = &ldsP[wid * 16 * 72];
  const int nkt = blockIdx.x * 2 + 2;           // causal: keys 0 .. qs+127

  for (int kt = 0; kt < nkt; ++kt) {
    const int key0 = kt * 64;
    __syncthreads();
    #pragma unroll
    for (int p = 0; p < 2; ++p) {               // cooperative K / V^T staging
      int chunk = p * 256 + tid;
      int row = chunk >> 3;
      int c8 = (chunk & 7) * 8;
      size_t g = base + (size_t)(key0 + row) * DH + c8;
      *(uint4*)&ldsK[row * 72 + c8] = *(const uint4*)&Kg[g];
      uint4 vv = *(const uint4*)&Vg[g];
      const unsigned short* vs = (const unsigned short*)&vv;
      #pragma unroll
      for (int j = 0; j < 8; ++j) ldsVT[(c8 + j) * 72 + row] = vs[j];
    }
    __syncthreads();

    // S = Q K^T  (Q already scaled by 1/sqrt(Dh))
    v8f s[4];
    #pragma unroll
    for (int j = 0; j < 4; ++j) s[j] = vzero8();
    #pragma unroll
    for (int c = 0; c < 2; ++c)
      #pragma unroll
      for (int j = 0; j < 4; ++j) {
        const unsigned short* pr = &ldsK[(j * 16 + lanen) * 72 + c * 32 + half * 8];
        v16bf bk = mk16(pr, pr + 16);
        s[j] = __builtin_amdgcn_wmma_f32_16x16x32_bf16(false, aq[c], false, bk, (short)0, s[j], false, false);
      }

    if (key0 + 63 > qs) {                       // diagonal tiles: causal mask (uniform branch)
      #pragma unroll
      for (int j = 0; j < 4; ++j) {
        int key = key0 + j * 16 + lanen;
        #pragma unroll
        for (int r = 0; r < 8; ++r) {
          int qq = qs + wid * 16 + r + half * 8;
          s[j][r] = (key > qq) ? -1e30f : s[j][r];
        }
      }
    }

    // online softmax per row (lanes 0-15 = rows r, lanes 16-31 = rows r+8)
    float psum[8];
    #pragma unroll
    for (int r = 0; r < 8; ++r) {
      float v = fmaxf(fmaxf(s[0][r], s[1][r]), fmaxf(s[2][r], s[3][r]));
      v = rmax16(v);                            // VALU permlane reduction (no DS traffic)
      float mnew = fmaxf(mrow[r], v);
      float sc = __expf(mrow[r] - mnew);
      mrow[r] = mnew;
      lrow[r] *= sc;
      #pragma unroll
      for (int d = 0; d < 4; ++d) o[d][r] *= sc;
      float p0 = __expf(s[0][r] - mnew), p1 = __expf(s[1][r] - mnew);
      float p2 = __expf(s[2][r] - mnew), p3 = __expf(s[3][r] - mnew);
      s[0][r] = p0; s[1][r] = p1; s[2][r] = p2; s[3][r] = p3;
      psum[r] = rsum16((p0 + p1) + (p2 + p3));
    }
    #pragma unroll
    for (int r = 0; r < 8; ++r) lrow[r] += psum[r];

    // transpose P (D-layout -> A-layout) through per-wave LDS
    #pragma unroll
    for (int j = 0; j < 4; ++j)
      #pragma unroll
      for (int r = 0; r < 8; ++r)
        pw[(r + half * 8) * 72 + j * 16 + lanen] = f2bf(s[j][r]);
    __asm__ volatile("s_wait_dscnt 0" ::: "memory");  // same-wave LDS RAW

    // O += P @ V
    #pragma unroll
    for (int c = 0; c < 2; ++c) {
      const unsigned short* pp = pw + lanen * 72 + c * 32 + half * 8;
      v16bf ap = mk16(pp, pp + 16);
      #pragma unroll
      for (int d = 0; d < 4; ++d) {
        const unsigned short* pv = &ldsVT[(d * 16 + lanen) * 72 + c * 32 + half * 8];
        v16bf bv = mk16(pv, pv + 16);
        o[d] = __builtin_amdgcn_wmma_f32_16x16x32_bf16(false, ap, false, bv, (short)0, o[d], false, false);
      }
    }
  }

  float inv[8];
  #pragma unroll
  for (int r = 0; r < 8; ++r) inv[r] = 1.0f / lrow[r];
  const int brow = bh / HEADS, h = bh % HEADS;
  #pragma unroll
  for (int d = 0; d < 4; ++d)
    #pragma unroll
    for (int r = 0; r < 8; ++r) {
      int trow = qs + wid * 16 + r + half * 8;
      Yg[(size_t)(brow * T_SEQ + trow) * CDIM + h * DH + d * 16 + lanen] = f2bf(o[d][r] * inv[r]);
    }
}

extern "C" void kernel_launch(void* const* d_in, const int* in_sizes, int n_in,
                              void* d_out, int out_size, void* d_ws, size_t ws_size,
                              hipStream_t stream) {
  const float* x      = (const float*)d_in[0];
  const float* W_attn = (const float*)d_in[1];
  const float* b_attn = (const float*)d_in[2];
  const float* W_proj = (const float*)d_in[3];
  const float* b_proj = (const float*)d_in[4];
  float* out = (float*)d_out;

  const int B = 2, C = CDIM;
  const size_t BT = (size_t)B * T_SEQ;            // 8192

  unsigned short* ws   = (unsigned short*)d_ws;
  unsigned short* xb   = ws;                       // [BT, C] bf16
  unsigned short* wabT = xb + BT * C;              // [3C, C] bf16 (W_attn^T)
  unsigned short* wpbT = wabT + (size_t)3 * C * C; // [C, C]  bf16 (W_proj^T)
  unsigned short* Qb   = wpbT + (size_t)C * C;     // [B,H,T,Dh] bf16
  unsigned short* Kb   = Qb + BT * C;
  unsigned short* Vb   = Kb + BT * C;
  unsigned short* Yb   = Vb + BT * C;              // [BT, C] bf16 attn output

  {
    int n = (int)(BT * C);
    convert_bf16<<<(n + 255) / 256, 256, 0, stream>>>(x, xb, n);
  }
  convert_transpose_bf16<<<(C * 3 * C + 255) / 256, 256, 0, stream>>>(W_attn, wabT, C, 3 * C);
  convert_transpose_bf16<<<(C * C + 255) / 256, 256, 0, stream>>>(W_proj, wpbT, C, C);

  dim3 g1(3 * C / 128, (int)(BT / 128));
  gemm_bf16<0><<<g1, BDIM, 0, stream>>>(xb, wabT, b_attn, nullptr, Qb, Kb, Vb,
                                        (int)BT, 3 * C, C);

  dim3 gf(T_SEQ / 128, B * HEADS);
  flash_attn<<<gf, BDIM, 0, stream>>>(Qb, Kb, Vb, Yb);

  dim3 g2(C / 128, (int)(BT / 128));
  gemm_bf16<1><<<g2, BDIM, 0, stream>>>(Yb, wpbT, b_proj, out, nullptr, nullptr, nullptr,
                                        (int)BT, C, C);
}
